// ModularGatedCascadeCondNet_76218489634818
// MI455X (gfx1250) — compile-verified, matrix operand order checked by reading, not run on hardware
//
#include <hip/hip_runtime.h>
#include <hip/hip_bf16.h>

// ---------------------------------------------------------------------------
// ModularGatedCascadeCondNet on MI455X (gfx1250, wave32, WMMA bf16)
//   L=4, M=4, DIN=128, DEM=64, DH=400, GH=256, DOUT=8, B=16384
// Strategy: all big GEMMs via v_wmma_f32_16x16x32_bf16 with
//   - weights pre-packed to per-lane-contiguous bf16 fragments (2x b128/wmma)
//   - BatchNorm folded into the A-operand as per-K-column scale/shift
//   - double-buffered LDS A tile (1 barrier / K-step), 2 N-tiles per wave
//   - deterministic block reductions for BN stats (no float atomics)
// ---------------------------------------------------------------------------

#define BATCH_DIV 128   // DIN
#define DH 400
#define GH 256
#define MM 4
#define DOUT 8
#define EPSBN 1e-5f

typedef __attribute__((ext_vector_type(16))) __bf16        v16bf;
typedef __attribute__((ext_vector_type(8)))  float         v8f;
typedef __attribute__((ext_vector_type(8)))  unsigned int  v8u;

__device__ __forceinline__ unsigned short f2bf(float f) {
    unsigned u = __builtin_bit_cast(unsigned, f);
    unsigned r = u + 0x7FFFu + ((u >> 16) & 1u);   // round-to-nearest-even
    return (unsigned short)(r >> 16);
}

__device__ __forceinline__ unsigned pack2bf(float lo, float hi) {
    return (unsigned)f2bf(lo) | ((unsigned)f2bf(hi) << 16);
}

__device__ __forceinline__ float wred(float v) {
    #pragma unroll
    for (int off = 16; off; off >>= 1) v += __shfl_xor(v, off, 32);
    return v;
}

// k index covered by fragment element e (0..15) for lane-half h (0/1).
// Matches ISA 7.12.2 "16-bit A-Matrix 16x32" (B assumed symmetric, lane=col).
__device__ __host__ __forceinline__ int kmap(int e, int h) {
    return ((e & 8) << 1) + (((e >> 1) & 3) << 1) + (e & 1) + (h << 3);
}

// ---------------------------------------------------------------------------
// Pack f32 [K,N] weight into bf16 WMMA-fragment order:
//   Wp[((kt*Nt + nt)*32 + lane)*16 + e] = W[kt*32 + kmap(e,lane>>4)][nt*16 + (lane&15)]
// OOB (K/N padding) -> 0, so the GEMM inner loop needs no guards on W.
// ---------------------------------------------------------------------------
__global__ __launch_bounds__(256) void pack_w_kernel(
    const float* __restrict__ W, int K, int N,
    unsigned short* __restrict__ Wp, int total)
{
    int idx = blockIdx.x * 256 + threadIdx.x;
    if (idx >= total) return;
    int Nt   = (N + 15) >> 4;
    int e    = idx & 15;
    int lane = (idx >> 4) & 31;
    int tile = idx >> 9;
    int kt = tile / Nt, nt = tile % Nt;
    int h = lane >> 4, c = lane & 15;
    int k = kt * 32 + kmap(e, h);
    int n = nt * 16 + c;
    float v = (k < K && n < N) ? W[(long)k * N + n] : 0.0f;
    Wp[idx] = f2bf(v);
}

// ---------------------------------------------------------------------------
// WMMA GEMM: C[z] = act( bn(A[z]) @ W[z] + bias[z] )
//   A: f32 [B,K] row-major (row stride lda), optional per-K scale/shift (BN)
//   Wp: packed bf16 fragments, NtW = ceil(N/16) tiles per K-step row
//   block = 256 thr = 8 waves (2 M x 4 N positions), each wave owns TWO
//   16-col tiles sharing one A fragment -> 32x128 C tile, 2 wmma / K-step.
//   LDS A tile double-buffered: one barrier per K-step.
// ---------------------------------------------------------------------------
template<int RELU>
__global__ __launch_bounds__(256) void gemm_kernel(
    const float* __restrict__ A, long lda, long sAz,
    const unsigned short* __restrict__ Wp, long sWz, int NtW,
    const float* __restrict__ bias, long sBz,
    const float* __restrict__ scale, const float* __restrict__ shift, long sSz,
    float* __restrict__ C, long ldc, long sCz,
    int K, int N)
{
    __shared__ unsigned short As[2][32][34];   // double-buffered 32x32 bf16 tile

    const int z = blockIdx.z;
    A  += (long)z * sAz;
    Wp += (long)z * sWz;
    C  += (long)z * sCz;
    const float* biasz  = bias  ? bias  + (long)z * sBz : nullptr;
    const float* scalez = scale ? scale + (long)z * sSz : nullptr;
    const float* shiftz = shift ? shift + (long)z * sSz : nullptr;

    const int tid  = threadIdx.x;
    const int lane = tid & 31;
    const int wave = tid >> 5;
    const int wm   = wave >> 2;        // 0..1  (16-row tile within block)
    const int wn   = wave & 3;         // 0..3  (pair of 16-col tiles)
    const int row0 = blockIdx.x * 32;
    const int ntile0 = blockIdx.y * 8 + wn * 2;   // global 16-col tile index
    const bool act0 = (ntile0 * 16) < N;
    const bool act1 = ((ntile0 + 1) * 16) < N;

    const int h  = lane >> 4;          // lane half
    const int cc = lane & 15;

    // cooperative A-tile staging: thread -> (row, 4 k-cols)
    const int ar  = tid >> 3;          // 0..31
    const int ac4 = (tid & 7) << 2;    // 0,4,...,28
    const float* aprow = A + (long)(row0 + ar) * lda;

    const int Kt = (K + 31) >> 5;

    v8f acc0, acc1;
    #pragma unroll
    for (int i = 0; i < 8; ++i) { acc0[i] = 0.0f; acc1[i] = 0.0f; }

    // stage K-step kt into LDS buffer buf (f32 -> BN -> bf16)
    auto stageA = [&](int buf, int kt) {
        const int k0 = kt << 5;
        if (k0 + 32 <= K) {            // interior tile: vector path
            float4 v = *(const float4*)(aprow + k0 + ac4);
            float x0 = v.x, x1 = v.y, x2 = v.z, x3 = v.w;
            if (scalez) {
                const int kb = k0 + ac4;
                x0 = x0 * scalez[kb+0] + shiftz[kb+0];
                x1 = x1 * scalez[kb+1] + shiftz[kb+1];
                x2 = x2 * scalez[kb+2] + shiftz[kb+2];
                x3 = x3 * scalez[kb+3] + shiftz[kb+3];
            }
            unsigned* dst = (unsigned*)&As[buf][ar][ac4];
            dst[0] = pack2bf(x0, x1);
            dst[1] = pack2bf(x2, x3);
        } else {                       // K-remainder tile: guarded path
            float xs[4];
            #pragma unroll
            for (int i = 0; i < 4; ++i) {
                int k = k0 + ac4 + i;
                float x = 0.0f;
                if (k < K) {
                    x = aprow[k];
                    if (scalez) x = x * scalez[k] + shiftz[k];
                }
                xs[i] = x;
            }
            unsigned* dst = (unsigned*)&As[buf][ar][ac4];
            dst[0] = pack2bf(xs[0], xs[1]);
            dst[1] = pack2bf(xs[2], xs[3]);
        }
    };

    auto loadB = [&](const unsigned short* p) -> v16bf {
        const uint4* wp = (const uint4*)p;
        uint4 q0 = wp[0], q1 = wp[1];
        v8u bu; bu[0]=q0.x; bu[1]=q0.y; bu[2]=q0.z; bu[3]=q0.w;
                bu[4]=q1.x; bu[5]=q1.y; bu[6]=q1.z; bu[7]=q1.w;
        return __builtin_bit_cast(v16bf, bu);
    };

    stageA(0, 0);
    __syncthreads();

    for (int kt = 0; kt < Kt; ++kt) {
        const int buf = kt & 1;
        if (kt + 1 < Kt) {
            stageA(buf ^ 1, kt + 1);
            // prefetch next K-step's B fragments (global_prefetch_b8)
            if (act0)
                __builtin_prefetch(Wp + (((long)(kt + 1) * NtW + ntile0) << 9) + lane * 16, 0, 3);
        }
        if (act0) {
            const unsigned short* wb = Wp + (((long)kt * NtW + ntile0) << 9) + lane * 16;
            v16bf bf0 = loadB(wb);
            // A fragment: 8 aligned b32 k-pair gathers from LDS
            const int r = wm * 16 + cc;
            v8u au;
            #pragma unroll
            for (int p = 0; p < 8; ++p) {
                int kp = ((p & 4) << 2) + ((p & 3) << 1) + (h << 3);
                au[p] = *(const unsigned int*)&As[buf][r][kp];
            }
            v16bf af = __builtin_bit_cast(v16bf, au);
            acc0 = __builtin_amdgcn_wmma_f32_16x16x32_bf16(
                       false, af, false, bf0, (short)0, acc0, false, false);
            if (act1) {
                v16bf bf1 = loadB(wb + 512);
                acc1 = __builtin_amdgcn_wmma_f32_16x16x32_bf16(
                           false, af, false, bf1, (short)0, acc1, false, false);
            }
        }
        __syncthreads();
    }

    // epilogue: bias + optional relu, documented C layout (vgpr i -> row i+8h)
    const int rbase = row0 + wm * 16 + (h << 3);
    if (act0) {
        const int colg = ntile0 * 16 + cc;
        if (colg < N) {
            float bb = biasz ? biasz[colg] : 0.0f;
            float* cp = C + (long)rbase * ldc + colg;
            #pragma unroll
            for (int i = 0; i < 8; ++i) {
                float v = acc0[i] + bb;
                if (RELU) v = fmaxf(v, 0.0f);
                cp[(long)i * ldc] = v;
            }
        }
    }
    if (act1) {
        const int colg = (ntile0 + 1) * 16 + cc;
        if (colg < N) {
            float bb = biasz ? biasz[colg] : 0.0f;
            float* cp = C + (long)rbase * ldc + colg;
            #pragma unroll
            for (int i = 0; i < 8; ++i) {
                float v = acc1[i] + bb;
                if (RELU) v = fmaxf(v, 0.0f);
                cp[(long)i * ldc] = v;
            }
        }
    }
}

// ---------------------------------------------------------------------------
// Column BN stats -> folded (scale, shift). One block per column, fixed-order
// tree reduction (deterministic). colmod!=0 replays a physical column for the
// broadcast-h layer-0 case.
// ---------------------------------------------------------------------------
__global__ __launch_bounds__(256) void bn_stats_kernel(
    const float* __restrict__ X, int ld, int colmod, int rows,
    const float* __restrict__ g, const float* __restrict__ b,
    float* __restrict__ scale, float* __restrict__ shift)
{
    __shared__ float ssum[256], ssq[256];
    const int c  = blockIdx.x;
    const int pc = colmod ? (c % colmod) : c;
    float s = 0.0f, s2 = 0.0f;
    for (int r = threadIdx.x; r < rows; r += 256) {
        float v = X[(long)r * ld + pc];
        s += v; s2 += v * v;
    }
    ssum[threadIdx.x] = s; ssq[threadIdx.x] = s2;
    __syncthreads();
    for (int off = 128; off; off >>= 1) {
        if (threadIdx.x < off) {
            ssum[threadIdx.x] += ssum[threadIdx.x + off];
            ssq[threadIdx.x]  += ssq[threadIdx.x + off];
        }
        __syncthreads();
    }
    if (threadIdx.x == 0) {
        float mu  = ssum[0] / rows;
        float var = ssq[0] / rows - mu * mu;
        if (var < 0.0f) var = 0.0f;
        float sc = g[c] * rsqrtf(var + EPSBN);
        scale[c] = sc;
        shift[c] = b[c] - mu * sc;
    }
}

// cond_ob = emb*out (then relu), out -> relu(out); both in place.
__global__ __launch_bounds__(256) void elemwise_cond_kernel(
    float* __restrict__ outp, float* __restrict__ emb0, int n)
{
    int i = blockIdx.x * 256 + threadIdx.x;
    if (i >= n) return;
    float o = outp[i], e = emb0[i];
    emb0[i] = fmaxf(e * o, 0.0f);
    outp[i] = fmaxf(o, 0.0f);
}

__device__ __forceinline__ void softmax4x4(float* v) {
    #pragma unroll
    for (int gidx = 0; gidx < 4; ++gidx) {
        float a = v[4*gidx], b = v[4*gidx+1], c = v[4*gidx+2], d = v[4*gidx+3];
        float m = fmaxf(fmaxf(a, b), fmaxf(c, d));
        float ea = expf(a - m), eb = expf(b - m), ec = expf(c - m), ed = expf(d - m);
        float r = 1.0f / (ea + eb + ec + ed);
        v[4*gidx] = ea*r; v[4*gidx+1] = eb*r; v[4*gidx+2] = ec*r; v[4*gidx+3] = ed*r;
    }
}

template<int T>
__device__ __forceinline__ void gate_stage(
    const float* flat, const float* __restrict__ condw, const float* __restrict__ condb,
    const float* __restrict__ gww, const float* __restrict__ gwb,
    const float* eg, int lane, float* raw)
{
    float val[8];
    #pragma unroll
    for (int j = 0; j < 8; ++j) {
        int c = lane + 32 * j;
        float cd = condb[c];
        #pragma unroll
        for (int t = 0; t < T; ++t) cd += flat[t] * condw[t * GH + c];
        val[j] = fmaxf(cd * eg[j], 0.0f);
    }
    #pragma unroll
    for (int o = 0; o < 16; ++o) {
        float p = 0.0f;
        #pragma unroll
        for (int j = 0; j < 8; ++j) p += val[j] * gww[(lane + 32*j) * 16 + o];
        raw[o] = wred(p) + gwb[o];
    }
}

// One wave32 per batch row: full gating cascade -> wAll[b,48], lw[b,4].
__global__ __launch_bounds__(256) void gating_kernel(
    const float* __restrict__ embG,
    const float* __restrict__ gw0_w,  const float* __restrict__ gw0_b,
    const float* __restrict__ cond1_w,const float* __restrict__ cond1_b,
    const float* __restrict__ gw1_w,  const float* __restrict__ gw1_b,
    const float* __restrict__ cond2_w,const float* __restrict__ cond2_b,
    const float* __restrict__ gw2_w,  const float* __restrict__ gw2_b,
    const float* __restrict__ condL_w,const float* __restrict__ condL_b,
    const float* __restrict__ gwL_w,  const float* __restrict__ gwL_b,
    float* __restrict__ wAll, float* __restrict__ lw, int Brows)
{
    const int wid = (blockIdx.x * 256 + threadIdx.x) >> 5;
    if (wid >= Brows) return;
    const int lane = threadIdx.x & 31;

    float eg[8], er[8];
    #pragma unroll
    for (int j = 0; j < 8; ++j) {
        eg[j] = embG[(long)wid * GH + lane + 32 * j];
        er[j] = fmaxf(eg[j], 0.0f);
    }
    float flat[48];
    float raw[16];
    // stage 0: relu(emb) @ gw0
    #pragma unroll
    for (int o = 0; o < 16; ++o) {
        float p = 0.0f;
        #pragma unroll
        for (int j = 0; j < 8; ++j) p += er[j] * gw0_w[(lane + 32*j) * 16 + o];
        raw[o] = wred(p) + gw0_b[o];
    }
    softmax4x4(raw);
    #pragma unroll
    for (int t = 0; t < 16; ++t) flat[t] = raw[t];
    // stage 1
    gate_stage<16>(flat, cond1_w, cond1_b, gw1_w, gw1_b, eg, lane, raw);
    softmax4x4(raw);
    #pragma unroll
    for (int t = 0; t < 16; ++t) flat[16 + t] = raw[t];
    // stage 2
    gate_stage<32>(flat, cond2_w, cond2_b, gw2_w, gw2_b, eg, lane, raw);
    softmax4x4(raw);
    #pragma unroll
    for (int t = 0; t < 16; ++t) flat[32 + t] = raw[t];
    // last-weight stage (4 outputs)
    float valL[8];
    #pragma unroll
    for (int j = 0; j < 8; ++j) {
        int c = lane + 32 * j;
        float cd = condL_b[c];
        #pragma unroll
        for (int t = 0; t < 48; ++t) cd += flat[t] * condL_w[t * GH + c];
        valL[j] = fmaxf(cd * eg[j], 0.0f);
    }
    float accL[4];
    #pragma unroll
    for (int o = 0; o < 4; ++o) {
        float p = 0.0f;
        #pragma unroll
        for (int j = 0; j < 8; ++j) p += valL[j] * gwL_w[(lane + 32*j) * 4 + o];
        accL[o] = wred(p) + gwL_b[o];
    }
    {   // softmax over 4
        float m = fmaxf(fmaxf(accL[0], accL[1]), fmaxf(accL[2], accL[3]));
        float e0 = expf(accL[0]-m), e1 = expf(accL[1]-m), e2 = expf(accL[2]-m), e3 = expf(accL[3]-m);
        float r = 1.0f / (e0 + e1 + e2 + e3);
        accL[0] = e0*r; accL[1] = e1*r; accL[2] = e2*r; accL[3] = e3*r;
    }
    if (lane == 0) {
        #pragma unroll
        for (int t = 0; t < 48; ++t) wAll[(long)wid * 48 + t] = flat[t];
        #pragma unroll
        for (int o = 0; o < 4; ++o)  lw[(long)wid * 4 + o] = accL[o];
    }
}

// mi[b,j,d] = relu( sum_k bn2(mo[b,k,d]) * w[b,j,k] )
__global__ __launch_bounds__(256) void dispatch_kernel(
    const float* __restrict__ mo, const float* __restrict__ s2, const float* __restrict__ h2,
    const float* __restrict__ wAll, int wOff, float* __restrict__ mi, int n)
{
    int i = blockIdx.x * 256 + threadIdx.x;
    if (i >= n) return;
    int b = i / DH, d = i % DH;
    float v[MM];
    #pragma unroll
    for (int k = 0; k < MM; ++k)
        v[k] = mo[(long)b * (MM*DH) + k * DH + d] * s2[k * DH + d] + h2[k * DH + d];
    const float* w = wAll + (long)b * 48 + wOff;
    #pragma unroll
    for (int j = 0; j < MM; ++j) {
        float a = v[0]*w[j*4+0] + v[1]*w[j*4+1] + v[2]*w[j*4+2] + v[3]*w[j*4+3];
        mi[(long)b * (MM*DH) + j * DH + d] = fmaxf(a, 0.0f);
    }
}

// out[b,:] = relu( sum_m bn2(mo[b,m,:]) * lw[b,m] ) @ last_w + last_b
__global__ __launch_bounds__(256) void final_kernel(
    const float* __restrict__ mo, const float* __restrict__ s2, const float* __restrict__ h2,
    const float* __restrict__ lw, const float* __restrict__ last_w, const float* __restrict__ last_b,
    float* __restrict__ out, int Brows)
{
    const int wid = (blockIdx.x * 256 + threadIdx.x) >> 5;
    if (wid >= Brows) return;
    const int lane = threadIdx.x & 31;
    float l0 = lw[(long)wid*4+0], l1 = lw[(long)wid*4+1];
    float l2 = lw[(long)wid*4+2], l3 = lw[(long)wid*4+3];
    float acc[DOUT];
    #pragma unroll
    for (int o = 0; o < DOUT; ++o) acc[o] = 0.0f;
    for (int d = lane; d < DH; d += 32) {
        const long base = (long)wid * (MM*DH) + d;
        float v = (mo[base + 0*DH]*s2[0*DH+d] + h2[0*DH+d]) * l0
                + (mo[base + 1*DH]*s2[1*DH+d] + h2[1*DH+d]) * l1
                + (mo[base + 2*DH]*s2[2*DH+d] + h2[2*DH+d]) * l2
                + (mo[base + 3*DH]*s2[3*DH+d] + h2[3*DH+d]) * l3;
        v = fmaxf(v, 0.0f);
        #pragma unroll
        for (int o = 0; o < DOUT; ++o) acc[o] += v * last_w[d * DOUT + o];
    }
    #pragma unroll
    for (int o = 0; o < DOUT; ++o) acc[o] = wred(acc[o]);
    if (lane == 0) {
        #pragma unroll
        for (int o = 0; o < DOUT; ++o) out[(long)wid * DOUT + o] = acc[o] + last_b[o];
    }
}

// ---------------------------------------------------------------------------
extern "C" void kernel_launch(void* const* d_in, const int* in_sizes, int n_in,
                              void* d_out, int out_size, void* d_ws, size_t ws_size,
                              hipStream_t stream) {
    const int B = in_sizes[0] / BATCH_DIV;   // 16384 (multiple of 32)

    const float* x       = (const float*)d_in[0];
    const float* embI    = (const float*)d_in[1];
    const float* base_w1 = (const float*)d_in[2];
    const float* base_b1 = (const float*)d_in[3];
    const float* base_w2 = (const float*)d_in[4];
    const float* base_b2 = (const float*)d_in[5];
    const float* em_w    = (const float*)d_in[6];
    const float* em_b    = (const float*)d_in[7];
    const float* gfc_w1  = (const float*)d_in[8];
    const float* gfc_b1  = (const float*)d_in[9];
    const float* gfc_w2  = (const float*)d_in[10];
    const float* gfc_b2  = (const float*)d_in[11];
    const float* gw0_w   = (const float*)d_in[12];
    const float* gw0_b   = (const float*)d_in[13];
    const float* cond1_w = (const float*)d_in[14];
    const float* cond1_b = (const float*)d_in[15];
    const float* gw1_w   = (const float*)d_in[16];
    const float* gw1_b   = (const float*)d_in[17];
    const float* cond2_w = (const float*)d_in[18];
    const float* cond2_b = (const float*)d_in[19];
    const float* gw2_w   = (const float*)d_in[20];
    const float* gw2_b   = (const float*)d_in[21];
    const float* condL_w = (const float*)d_in[22];
    const float* condL_b = (const float*)d_in[23];
    const float* gwL_w   = (const float*)d_in[24];
    const float* gwL_b   = (const float*)d_in[25];
    const float* bn1_g   = (const float*)d_in[26];
    const float* bn1_b   = (const float*)d_in[27];
    const float* mod_W   = (const float*)d_in[28];
    const float* mod_b   = (const float*)d_in[29];
    const float* bn2_g   = (const float*)d_in[30];
    const float* bn2_b   = (const float*)d_in[31];
    const float* last_w  = (const float*)d_in[32];
    const float* last_b  = (const float*)d_in[33];

    // ---- workspace carve-out ----
    char* ws = (char*)d_ws;
    size_t off = 0;
    auto carve = [&](size_t bytes) -> void* {
        void* p = ws + off;
        off = (off + bytes + 255) & ~(size_t)255;
        return p;
    };
    const size_t P400 = 13 * 25 * 512;                   // packed size K=400,N=400
    unsigned short* P_bw1 = (unsigned short*)carve(2 * (size_t)(4  * 25 * 512));
    unsigned short* P_bw2 = (unsigned short*)carve(2 * P400);
    unsigned short* P_em  = (unsigned short*)carve(2 * (size_t)(2  * 25 * 512));
    unsigned short* P_g1  = (unsigned short*)carve(2 * (size_t)(13 * 16 * 512));
    unsigned short* P_g2  = (unsigned short*)carve(2 * (size_t)(8  * 16 * 512));
    unsigned short* P_mod = (unsigned short*)carve(2 * 16 * P400);
    float* out1  = (float*)carve((size_t)B * DH * 4);
    float* outp  = (float*)carve((size_t)B * DH * 4);    // out (pre/post relu, in place)
    float* emb0  = (float*)carve((size_t)B * DH * 4);    // emb / relu(emb*out), in place
    float* g1buf = (float*)carve((size_t)B * GH * 4);
    float* embG  = (float*)carve((size_t)B * GH * 4);
    float* wAll  = (float*)carve((size_t)B * 48 * 4);
    float* lwbuf = (float*)carve((size_t)B * 4 * 4);
    float* mibuf = (float*)carve((size_t)B * MM * DH * 4);
    float* mobuf = (float*)carve((size_t)B * MM * DH * 4);
    float* s1 = (float*)carve(MM * DH * 4);
    float* h1 = (float*)carve(MM * DH * 4);
    float* s2 = (float*)carve(MM * DH * 4);
    float* h2 = (float*)carve(MM * DH * 4);
    (void)ws_size; (void)n_in; (void)out_size;

    auto pack = [&](const float* W, int K, int N, unsigned short* Wp) {
        int total = ((K + 31) / 32) * ((N + 15) / 16) * 512;
        pack_w_kernel<<<(total + 255) / 256, 256, 0, stream>>>(W, K, N, Wp, total);
    };
    auto gemm = [&](bool relu, const float* A, long lda, long sAz,
                    const unsigned short* Wp, long sWz, int NtW,
                    const float* bias, long sBz,
                    const float* sc, const float* sh, long sSz,
                    float* C, long ldc, long sCz, int K, int N, int Z) {
        int Nt = (N + 15) / 16;
        dim3 g(B / 32, (Nt + 7) / 8, Z);
        if (relu)
            gemm_kernel<1><<<g, 256, 0, stream>>>(A, lda, sAz, Wp, sWz, NtW,
                                                  bias, sBz, sc, sh, sSz, C, ldc, sCz, K, N);
        else
            gemm_kernel<0><<<g, 256, 0, stream>>>(A, lda, sAz, Wp, sWz, NtW,
                                                  bias, sBz, sc, sh, sSz, C, ldc, sCz, K, N);
    };

    // ---- pack all GEMM weights to bf16 fragments (once per launch) ----
    pack(base_w1, 128, DH, P_bw1);
    pack(base_w2, DH,  DH, P_bw2);
    pack(em_w,    64,  DH, P_em);
    pack(gfc_w1,  DH,  GH, P_g1);
    pack(gfc_w2,  GH,  GH, P_g2);
    for (int i = 0; i < 16; ++i)
        pack(mod_W + (size_t)i * DH * DH, DH, DH, P_mod + (size_t)i * P400);

    // ---- front-end ----
    gemm(true,  x,     128, 0, P_bw1, 0, 25, base_b1, 0, nullptr, nullptr, 0, out1, DH, 0, 128, DH, 1);
    gemm(false, out1,  DH,  0, P_bw2, 0, 25, base_b2, 0, nullptr, nullptr, 0, outp, DH, 0, DH,  DH, 1);
    gemm(false, embI,  64,  0, P_em,  0, 25, em_b,    0, nullptr, nullptr, 0, emb0, DH, 0, 64,  DH, 1);
    elemwise_cond_kernel<<<((size_t)B * DH + 255) / 256, 256, 0, stream>>>(outp, emb0, B * DH);
    gemm(true,  emb0,  DH,  0, P_g1,  0, 16, gfc_b1,  0, nullptr, nullptr, 0, g1buf, GH, 0, DH, GH, 1);
    gemm(false, g1buf, GH,  0, P_g2,  0, 16, gfc_b2,  0, nullptr, nullptr, 0, embG,  GH, 0, GH, GH, 1);

    // ---- gating cascade (one wave32 per row) ----
    gating_kernel<<<(B + 7) / 8, 256, 0, stream>>>(
        embG, gw0_w, gw0_b, cond1_w, cond1_b, gw1_w, gw1_b,
        cond2_w, cond2_b, gw2_w, gw2_b, condL_w, condL_b, gwL_w, gwL_b,
        wAll, lwbuf, B);

    // ---- module layer 0 (broadcast input, BN1 folded into A) ----
    bn_stats_kernel<<<MM * DH, 256, 0, stream>>>(outp, DH, DH, B, bn1_g, bn1_b, s1, h1);
    gemm(false, outp, DH, 0, P_mod, P400, 25, mod_b, DH, s1, h1, DH,
         mobuf, MM * DH, DH, DH, DH, MM);

    // ---- cascaded module layers ----
    for (int i = 1; i < 4; ++i) {
        bn_stats_kernel<<<MM * DH, 256, 0, stream>>>(mobuf, MM * DH, 0, B,
            bn2_g + (i - 1) * MM * DH, bn2_b + (i - 1) * MM * DH, s2, h2);
        dispatch_kernel<<<((size_t)B * DH + 255) / 256, 256, 0, stream>>>(
            mobuf, s2, h2, wAll, (i - 1) * 16, mibuf, B * DH);
        bn_stats_kernel<<<MM * DH, 256, 0, stream>>>(mibuf, MM * DH, 0, B,
            bn1_g + i * MM * DH, bn1_b + i * MM * DH, s1, h1);
        gemm(false, mibuf, MM * DH, DH, P_mod + (size_t)i * MM * P400, P400, 25,
             mod_b + i * MM * DH, DH, s1, h1, DH,
             mobuf, MM * DH, DH, DH, DH, MM);
    }

    // ---- final BN2 + weighted combine + last linear ----
    bn_stats_kernel<<<MM * DH, 256, 0, stream>>>(mobuf, MM * DH, 0, B,
        bn2_g + 3 * MM * DH, bn2_b + 3 * MM * DH, s2, h2);
    final_kernel<<<(B + 7) / 8, 256, 0, stream>>>(
        mobuf, s2, h2, lwbuf, last_w, last_b, (float*)d_out, B);
}
